// Transformer_59459527246629
// MI455X (gfx1250) — matrix operand order repaired and checked by reading, not compiled, not run on hardware
//
#include <hip/hip_runtime.h>
#include <math.h>

// ---------------------------------------------------------------------------
// Model constants (from the reference)
// ---------------------------------------------------------------------------
constexpr int cB = 2, cSC = 256, cST = 512, cD = 128, cDFF = 512, cH = 4;
constexpr int cNB = 4, cELEM = 50, cNBOND = 8;

// ---------------------------------------------------------------------------
// WMMA types / helpers (gfx1250, wave32)
// ---------------------------------------------------------------------------
typedef __attribute__((ext_vector_type(16))) __bf16 v16bf;
typedef __attribute__((ext_vector_type(8)))  float  v8f;

// Native f32 -> bf16 convert (v_cvt_pk_bf16_f32 on gfx1250, RNE).
static __device__ inline __bf16 f2bf(float f) { return (__bf16)f; }

// A fragment: 16x32 bf16, row-major fp32 source [lda]. contiguous k order used by this code (not the ISA table order):
// lanes 0-15 rows M=0..15 hold K = hi*16 + {0..15}
static __device__ inline v16bf load_a16x32(const float* A, int lda, int m0,
                                           int kk, int lane) {
  int r = lane & 15, hi = lane >> 4;
  const float* p = A + (long long)(m0 + r) * lda + kk + hi * 16;
  float4 f0 = *(const float4*)(p);
  float4 f1 = *(const float4*)(p + 4);
  float4 f2 = *(const float4*)(p + 8);
  float4 f3 = *(const float4*)(p + 12);
  v16bf a;
  a[0] = f2bf(f0.x);  a[1] = f2bf(f0.y);  a[2]  = f2bf(f0.z);  a[3]  = f2bf(f0.w);
  a[4] = f2bf(f1.x);  a[5] = f2bf(f1.y);  a[6]  = f2bf(f1.z);  a[7]  = f2bf(f1.w);
  a[8] = f2bf(f2.x);  a[9] = f2bf(f2.y);  a[10] = f2bf(f2.z);  a[11] = f2bf(f2.w);
  a[12] = f2bf(f3.x); a[13] = f2bf(f3.y); a[14] = f2bf(f3.z);  a[15] = f2bf(f3.w);
  return a;
}

// B fragment, B in memory row-major [K, N] (nn). Lane holds column n0+(lane&15),
// K = hi*16 + e  (e = element index 0..15) -> strided-by-ldb scalar loads.
// Column clamped to N-1 so edge tiles stay in-bounds (results masked on store).
static __device__ inline v16bf load_bn16x32(const float* Bm, int ldb, int n0,
                                            int kk, int lane, int N) {
  int c = n0 + (lane & 15);
  if (c >= N) c = N - 1;
  int hi = lane >> 4;
  const float* p = Bm + (long long)(kk + hi * 16) * ldb + c;
  v16bf b;
#pragma unroll
  for (int e = 0; e < 16; ++e) b[e] = f2bf(p[(long long)e * ldb]);
  return b;
}

// B fragment, B in memory row-major [N, K] (nt, i.e. B^T): contiguous K run.
static __device__ inline v16bf load_bt16x32(const float* Bm, int ldb, int n0,
                                            int kk, int lane, int N) {
  int c = n0 + (lane & 15);
  if (c >= N) c = N - 1;
  int hi = lane >> 4;
  const float* p = Bm + (long long)c * ldb + kk + hi * 16;
  float4 f0 = *(const float4*)(p);
  float4 f1 = *(const float4*)(p + 4);
  float4 f2 = *(const float4*)(p + 8);
  float4 f3 = *(const float4*)(p + 12);
  v16bf b;
  b[0] = f2bf(f0.x);  b[1] = f2bf(f0.y);  b[2]  = f2bf(f0.z);  b[3]  = f2bf(f0.w);
  b[4] = f2bf(f1.x);  b[5] = f2bf(f1.y);  b[6]  = f2bf(f1.z);  b[7]  = f2bf(f1.w);
  b[8] = f2bf(f2.x);  b[9] = f2bf(f2.y);  b[10] = f2bf(f2.z);  b[11] = f2bf(f2.w);
  b[12] = f2bf(f3.x); b[13] = f2bf(f3.y); b[14] = f2bf(f3.z);  b[15] = f2bf(f3.w);
  return b;
}

// ---------------------------------------------------------------------------
// Generic batched WMMA GEMM: C = act(A @ B (+bias)).
// One wave computes a 16x64 tile (4 accumulators); A fragment loaded once per
// K-step and reused across 4 wmma ops. The K-loop bodies are branch-free and
// compute all 4 tiles unconditionally (B columns clamped, stores guarded), so
// the 4 independent wmma chains can be interleaved to hide the RAW hazard.
// M must be a multiple of 16, K a multiple of 32 (true for every GEMM here).
// ---------------------------------------------------------------------------
__global__ __launch_bounds__(32) void k_gemm(
    const float* __restrict__ A, long long sA, int lda,
    const float* __restrict__ Bm, long long sB, int ldb, int bt,
    const float* __restrict__ bias, long long sBias,
    float* __restrict__ C, long long sC, int ldc,
    int M, int N, int K, int relu) {
  int lane = threadIdx.x;
  int m0 = blockIdx.x * 16, n0 = blockIdx.y * 64, z = blockIdx.z;
  A += (long long)z * sA;
  Bm += (long long)z * sB;
  C += (long long)z * sC;
  if (bias) bias += (long long)z * sBias;

  v8f acc[4] = {};
  if (bt) {
    for (int kk = 0; kk < K; kk += 32) {
      v16bf a = load_a16x32(A, lda, m0, kk, lane);
#pragma unroll
      for (int t = 0; t < 4; ++t) {
        v16bf b = load_bt16x32(Bm, ldb, n0 + 16 * t, kk, lane, N);
        acc[t] = __builtin_amdgcn_wmma_f32_16x16x32_bf16(false, a, false, b,
                                                         (short)0, acc[t],
                                                         false, false);
      }
    }
  } else {
    for (int kk = 0; kk < K; kk += 32) {
      v16bf a = load_a16x32(A, lda, m0, kk, lane);
#pragma unroll
      for (int t = 0; t < 4; ++t) {
        v16bf b = load_bn16x32(Bm, ldb, n0 + 16 * t, kk, lane, N);
        acc[t] = __builtin_amdgcn_wmma_f32_16x16x32_bf16(false, a, false, b,
                                                         (short)0, acc[t],
                                                         false, false);
      }
    }
  }

  int cl = lane & 15, hi = lane >> 4;
#pragma unroll
  for (int t = 0; t < 4; ++t) {
    int c = n0 + 16 * t + cl;
    if (c < N) {
      float bv = bias ? bias[c] : 0.f;
#pragma unroll
      for (int v = 0; v < 8; ++v) {
        int row = m0 + hi * 8 + v;                 // C/D layout: M = v + 8*hi
        float val = acc[t][v] + bv;
        if (relu) val = fmaxf(val, 0.f);
        C[(long long)row * ldc + c] = val;
      }
    }
  }
}

// ---------------------------------------------------------------------------
// Embedding gathers
// ---------------------------------------------------------------------------
__global__ __launch_bounds__(128) void k_embed3(
    const int* __restrict__ E, const int* __restrict__ Cg,
    const float* __restrict__ P, const float* __restrict__ eemb,
    const float* __restrict__ cemb, const float* __restrict__ posW,
    const float* __restrict__ posb, float* __restrict__ feat) {
  int r = blockIdx.x, t = threadIdx.x;
  int e = E[r], c = Cg[r];
  float p0 = P[r * 3], p1 = P[r * 3 + 1], p2 = P[r * 3 + 2];
  long long base = (long long)r * (3 * cD);
  feat[base + t] = eemb[(long long)e * cD + t];
  feat[base + cD + t] = cemb[(long long)c * cD + t];
  float pv = p0 * posW[t] + p1 * posW[cD + t] + p2 * posW[2 * cD + t] + posb[t];
  feat[base + 2 * cD + t] = fmaxf(pv, 0.f);
}

__global__ __launch_bounds__(128) void k_embed2(
    const int* __restrict__ E, const float* __restrict__ P,
    const float* __restrict__ eemb, const float* __restrict__ posW,
    const float* __restrict__ posb, float* __restrict__ feat) {
  int r = blockIdx.x, t = threadIdx.x;
  int e = E[r];
  float p0 = P[r * 3], p1 = P[r * 3 + 1], p2 = P[r * 3 + 2];
  long long base = (long long)r * (2 * cD);
  feat[base + t] = eemb[(long long)e * cD + t];
  float pv = p0 * posW[t] + p1 * posW[cD + t] + p2 * posW[2 * cD + t] + posb[t];
  feat[base + cD + t] = fmaxf(pv, 0.f);
}

// concat([rx,px], -1) -> [B*SC, 2D]
__global__ void k_cat(const float* __restrict__ rx, const float* __restrict__ px,
                      float* __restrict__ cat, int total) {
  for (int idx = blockIdx.x * blockDim.x + threadIdx.x; idx < total;
       idx += gridDim.x * blockDim.x) {
    int c = idx % (2 * cD), r = idx / (2 * cD);
    cat[idx] = (c < cD) ? rx[(long long)r * cD + c]
                        : px[(long long)r * cD + c - cD];
  }
}

// [H,B,S,D] -> [B*S, H*D] (head concat, transpose (1,2,0,3))
__global__ void k_repack(const float* __restrict__ o, float* __restrict__ oc,
                         int S, int total) {
  for (int idx = blockIdx.x * blockDim.x + threadIdx.x; idx < total;
       idx += gridDim.x * blockDim.x) {
    int d = idx % cD, t = idx / cD;
    int s = t % S; t /= S;
    int b = t % cB, h = t / cB;
    oc[((long long)(b * S + s)) * (cH * cD) + h * cD + d] = o[idx];
  }
}

// ---------------------------------------------------------------------------
// Score post-process + softmax over keys.
// Folds wsum = bond_emb.sum(-1)[ADJ] (no materialized [B,S,S] buffer),
// optional causal mask, 1/sqrt(Sk) scale.
// ---------------------------------------------------------------------------
__global__ __launch_bounds__(256) void k_smax(
    float* __restrict__ s, int Sq, int Sk, float scale,
    const int* __restrict__ adj, const float* __restrict__ bemb, int causal) {
  __shared__ float red[256];
  __shared__ float bsum[cNBOND];
  int t = threadIdx.x, r = blockIdx.x;
  int i = r % Sq;
  int b = (r / Sq) % cB;
  float* row = s + (long long)r * Sk;
  const int* arow = adj ? adj + ((long long)b * Sq + i) * Sk : nullptr;
  if (adj && t < cNBOND) {
    float sm = 0.f;
    for (int d = 0; d < cD; ++d) sm += bemb[t * cD + d];
    bsum[t] = sm;
  }
  __syncthreads();

  float lmax = -__builtin_inff();
  for (int j = t; j < Sk; j += 256) {
    float v = row[j] * scale;
    if (arow) v *= bsum[arow[j]];
    if (causal && j > i) v = -__builtin_inff();
    row[j] = v;
    lmax = fmaxf(lmax, v);
  }
  red[t] = lmax; __syncthreads();
  for (int o = 128; o; o >>= 1) { if (t < o) red[t] = fmaxf(red[t], red[t + o]); __syncthreads(); }
  float mx = red[0]; __syncthreads();

  float ls = 0.f;
  for (int j = t; j < Sk; j += 256) {
    float e = expf(row[j] - mx);
    row[j] = e; ls += e;
  }
  red[t] = ls; __syncthreads();
  for (int o = 128; o; o >>= 1) { if (t < o) red[t] += red[t + o]; __syncthreads(); }
  float inv = 1.f / red[0];
  for (int j = t; j < Sk; j += 256) row[j] *= inv;
}

// ---------------------------------------------------------------------------
// x = LayerNorm(a + x) * g + be   (in-place on x; one 128-thread block per row)
// ---------------------------------------------------------------------------
__global__ __launch_bounds__(128) void k_addln(const float* __restrict__ a,
                                               const float* __restrict__ g,
                                               const float* __restrict__ be,
                                               float* __restrict__ x) {
  __shared__ float red[cD];
  int r = blockIdx.x, t = threadIdx.x;
  long long idx = (long long)r * cD + t;
  float v = a[idx] + x[idx];
  red[t] = v; __syncthreads();
  for (int s = 64; s; s >>= 1) { if (t < s) red[t] += red[t + s]; __syncthreads(); }
  float mean = red[0] / cD; __syncthreads();
  float d = v - mean;
  red[t] = d * d; __syncthreads();
  for (int s = 64; s; s >>= 1) { if (t < s) red[t] += red[t + s]; __syncthreads(); }
  float var = red[0] / cD;
  x[idx] = g[t] * d * rsqrtf(var + 1e-5f) + be[t];
}

// ---------------------------------------------------------------------------
// Fused bond head (never materializes the 268MB `up` tensor):
// per (b, i-tile, j): up16x128 = relu(tx_tile @ b1W[:, j*128:+128] + b1b)  (WMMA)
//                     h16x128  = relu(up @ b2W1 + b2b1)                   (WMMA via LDS)
//                     hb16x8   = h @ b2W2 + b2b2                          (dot)
// A fragment loaded once per K-step, reused across all 8 N-tiles.
// ---------------------------------------------------------------------------
__global__ __launch_bounds__(32) void k_bond(
    const float* __restrict__ tx, const float* __restrict__ b1W,
    const float* __restrict__ b1b, const float* __restrict__ b2W1,
    const float* __restrict__ b2b1, const float* __restrict__ b2W2,
    const float* __restrict__ b2b2, float* __restrict__ hb) {
  __shared__ __align__(16) float sUp[16][132];
  __shared__ __align__(16) float sH[16][132];
  int lane = threadIdx.x;
  int i0 = blockIdx.x * 16, j = blockIdx.y, b = blockIdx.z;
  int hi = lane >> 4, cl = lane & 15;
  const float* A = tx + (long long)b * cST * cD;

  // GEMM1: up tile -> LDS (bias + relu)
  {
    v8f acc[8] = {};
#pragma unroll
    for (int kk = 0; kk < cD; kk += 32) {
      v16bf a = load_a16x32(A, cD, i0, kk, lane);
#pragma unroll
      for (int nt = 0; nt < 8; ++nt) {
        v16bf bm = load_bn16x32(b1W + (long long)j * cD, cD * cST,
                                nt * 16, kk, lane, cD);
        acc[nt] = __builtin_amdgcn_wmma_f32_16x16x32_bf16(false, a, false, bm,
                                                          (short)0, acc[nt],
                                                          false, false);
      }
    }
#pragma unroll
    for (int nt = 0; nt < 8; ++nt) {
      int c = nt * 16 + cl;
      float bv = b1b[(long long)j * cD + c];
#pragma unroll
      for (int v = 0; v < 8; ++v) sUp[hi * 8 + v][c] = fmaxf(acc[nt][v] + bv, 0.f);
    }
  }
  __syncthreads();

  // GEMM2: h = relu(up @ b2W1 + b2b1) -> LDS
  {
    v8f acc[8] = {};
#pragma unroll
    for (int kk = 0; kk < cD; kk += 32) {
      v16bf a = load_a16x32(&sUp[0][0], 132, 0, kk, lane);
#pragma unroll
      for (int nt = 0; nt < 8; ++nt) {
        v16bf bm = load_bn16x32(b2W1, cD, nt * 16, kk, lane, cD);
        acc[nt] = __builtin_amdgcn_wmma_f32_16x16x32_bf16(false, a, false, bm,
                                                          (short)0, acc[nt],
                                                          false, false);
      }
    }
#pragma unroll
    for (int nt = 0; nt < 8; ++nt) {
      int c = nt * 16 + cl;
      float bv = b2b1[c];
#pragma unroll
      for (int v = 0; v < 8; ++v) sH[hi * 8 + v][c] = fmaxf(acc[nt][v] + bv, 0.f);
    }
  }
  __syncthreads();

  // GEMM3 (N=8): scalar dots, 4 outputs/lane
#pragma unroll
  for (int t = 0; t < 4; ++t) {
    int idx = lane * 4 + t;
    int i = idx >> 3, n = idx & 7;
    float s = b2b2[n];
    for (int k = 0; k < cD; ++k) s += sH[i][k] * b2W2[k * cNBOND + n];
    hb[(((long long)b * cST + (i0 + i)) * cST + j) * cNBOND + n] = s;
  }
}

// bond softmax over axis=1 (i), in-place; one wave per (b,j,n) column
__global__ __launch_bounds__(32) void k_bond_softmax(float* __restrict__ hb) {
  int lane = threadIdx.x, c = blockIdx.x;
  int n = c & 7, jc = c >> 3;
  int j = jc & (cST - 1), b = jc >> 9;
  float* base = hb + ((long long)b * cST * cST + j) * cNBOND + n;
  const long long stride = (long long)cST * cNBOND;
  float ev[cST / 32];
  float mx = -__builtin_inff();
  for (int i = lane, t = 0; i < cST; i += 32, ++t) {
    float v = base[(long long)i * stride];
    ev[t] = v;
    mx = fmaxf(mx, v);
  }
  for (int off = 16; off; off >>= 1) mx = fmaxf(mx, __shfl_xor(mx, off));
  float sum = 0.f;
#pragma unroll
  for (int t = 0; t < cST / 32; ++t) { ev[t] = expf(ev[t] - mx); sum += ev[t]; }
  for (int off = 16; off; off >>= 1) sum += __shfl_xor(sum, off);
  float inv = 1.f / sum;
  for (int i = lane, t = 0; i < cST; i += 32, ++t)
    base[(long long)i * stride] = ev[t] * inv;
}

// elem softmax over axis=0 (batch, B=2)
__global__ void k_elem_softmax(const float* __restrict__ he, float* __restrict__ out) {
  int idx = blockIdx.x * blockDim.x + threadIdx.x;
  if (idx >= cST * cELEM) return;
  long long o0 = idx, o1 = (long long)cST * cELEM + idx;
  float v0 = he[o0], v1 = he[o1];
  float m = fmaxf(v0, v1);
  float e0 = expf(v0 - m), e1 = expf(v1 - m);
  float inv = 1.f / (e0 + e1);
  out[o0] = e0 * inv;
  out[o1] = e1 * inv;
}

// ---------------------------------------------------------------------------
// Host glue
// ---------------------------------------------------------------------------
struct GA  { const float *Wq,*bq,*Wk,*bk,*Wv,*bv,*Wo,*bo; };
struct Comp{ const float *elem,*charge,*bond,*posW,*posb,*ffW,*ffb; GA ga; };
struct Theo{ const float *elem,*bond,*posW,*posb,*ffW,*ffb; GA ga; };
struct Enc { GA mha; const float *W1,*b1,*W2,*b2,*g,*be; };
struct Dec { GA smha,cmha; const float *W1,*b1,*W2,*b2,*g,*be; };
struct Out { const float *eW1,*eb1,*eW2,*eb2,*pW1,*pb1,*pW2,*pb2,
                         *b1W,*b1b,*b2W1,*b2b1,*b2W2,*b2b2; };

extern "C" void kernel_launch(void* const* d_in, const int* in_sizes, int n_in,
                              void* d_out, int out_size, void* d_ws, size_t ws_size,
                              hipStream_t stream) {
  (void)in_sizes; (void)n_in; (void)out_size; (void)ws_size;

  // ---- inputs (setup_inputs() insertion order) ----
  const int*   RE   = (const int*)  d_in[0];
  const int*   RC   = (const int*)  d_in[1];
  const float* RP   = (const float*)d_in[2];
  const int*   RADJ = (const int*)  d_in[3];
  const int*   PE   = (const int*)  d_in[4];
  const int*   PC   = (const int*)  d_in[5];
  const float* PP   = (const float*)d_in[6];
  const int*   PADJ = (const int*)  d_in[7];
  const int*   TE   = (const int*)  d_in[8];
  const float* TP   = (const float*)d_in[9];
  const int*   TADJ = (const int*)  d_in[10];

  // ---- params: assume pytree flattened in nested-dict INSERTION order ----
  int ix = 11;
  auto FP = [&](void) { return (const float*)d_in[ix++]; };
  auto getGA = [&](void) { GA g; g.Wq=FP(); g.bq=FP(); g.Wk=FP(); g.bk=FP();
                           g.Wv=FP(); g.bv=FP(); g.Wo=FP(); g.bo=FP(); return g; };
  auto getComp = [&](void) { Comp c; c.elem=FP(); c.charge=FP(); c.bond=FP();
                             c.posW=FP(); c.posb=FP(); c.ffW=FP(); c.ffb=FP();
                             c.ga=getGA(); return c; };
  Comp react = getComp();
  Comp prod  = getComp();
  const float* cW = FP(); const float* cb = FP();
  Theo theo; theo.elem=FP(); theo.bond=FP(); theo.posW=FP(); theo.posb=FP();
  theo.ffW=FP(); theo.ffb=FP(); theo.ga=getGA();
  Enc enc[cNB];
  for (int i = 0; i < cNB; ++i) { enc[i].mha=getGA(); enc[i].W1=FP(); enc[i].b1=FP();
    enc[i].W2=FP(); enc[i].b2=FP(); enc[i].g=FP(); enc[i].be=FP(); }
  Dec dec[cNB];
  for (int i = 0; i < cNB; ++i) { dec[i].smha=getGA(); dec[i].cmha=getGA();
    dec[i].W1=FP(); dec[i].b1=FP(); dec[i].W2=FP(); dec[i].b2=FP();
    dec[i].g=FP(); dec[i].be=FP(); }
  Out oo; oo.eW1=FP(); oo.eb1=FP(); oo.eW2=FP(); oo.eb2=FP();
  oo.pW1=FP(); oo.pb1=FP(); oo.pW2=FP(); oo.pb2=FP();
  oo.b1W=FP(); oo.b1b=FP(); oo.b2W1=FP(); oo.b2b1=FP(); oo.b2W2=FP(); oo.b2b2=FP();

  // ---- workspace (bump allocator over d_ws, ~26 MB total) ----
  float* wsf = (float*)d_ws;
  size_t off = 0;
  auto alloc = [&](size_t n) { float* p = wsf + off; off += (n + 63) & ~(size_t)63; return p; };
  float* feat = alloc((size_t)cB * cST * 2 * cD);       // max of 512*384 / 1024*256
  float* cbuf = alloc((size_t)cB * cST * cD);
  float* rx   = alloc((size_t)cB * cSC * cD);
  float* px   = alloc((size_t)cB * cSC * cD);
  float* cat  = alloc((size_t)cB * cSC * 2 * cD);
  float* xb   = alloc((size_t)cB * cSC * cD);
  float* txb  = alloc((size_t)cB * cST * cD);
  float* qb   = alloc((size_t)cH * cB * cST * cD);
  float* kb   = alloc((size_t)cH * cB * cST * cD);
  float* vb   = alloc((size_t)cH * cB * cST * cD);
  float* sc   = alloc((size_t)cH * cB * cST * cST);
  float* ob   = alloc((size_t)cH * cB * cST * cD);
  float* ocb  = alloc((size_t)cB * cST * cH * cD);
  float* abuf = alloc((size_t)cB * cST * cD);
  float* ffh  = alloc((size_t)cB * cST * cDFF);
  float* t1   = alloc((size_t)cB * cST * cD);
  float* heb  = alloc((size_t)cB * cST * cELEM);

  float* out_elem = (float*)d_out;
  float* out_pos  = out_elem + (size_t)cB * cST * cELEM;
  float* out_bond = out_pos  + (size_t)cB * cST * 3;

  auto gemm = [&](const float* A, long long sA, int lda,
                  const float* Bm, long long sB, int ldb, int bt,
                  const float* bias, long long sBias,
                  float* C, long long sC, int ldc,
                  int M, int N, int K, int relu, int batches) {
    dim3 grid(M / 16, (N + 63) / 64, batches);
    k_gemm<<<grid, 32, 0, stream>>>(A, sA, lda, Bm, sB, ldb, bt, bias, sBias,
                                    C, sC, ldc, M, N, K, relu);
  };

  auto mha = [&](const GA& g, const float* xq, int Sq, const float* xkv, int Sk,
                 const int* adj, const float* bemb, int causal, float* outp) {
    int Mq = cB * Sq, Mk = cB * Sk;
    // per-head projections: q/k/v [H, B*S, D]
    gemm(xq, 0, cD, g.Wq, (long long)cD * cD, cD, 0, g.bq, cD,
         qb, (long long)Mq * cD, cD, Mq, cD, cD, 0, cH);
    gemm(xkv, 0, cD, g.Wk, (long long)cD * cD, cD, 0, g.bk, cD,
         kb, (long long)Mk * cD, cD, Mk, cD, cD, 0, cH);
    gemm(xkv, 0, cD, g.Wv, (long long)cD * cD, cD, 0, g.bv, cD,
         vb, (long long)Mk * cD, cD, Mk, cD, cD, 0, cH);
    // scores = q @ k^T  (NT), batched over H*B
    gemm(qb, (long long)Sq * cD, cD, kb, (long long)Sk * cD, cD, 1, nullptr, 0,
         sc, (long long)Sq * Sk, Sk, Sq, Sk, cD, 0, cH * cB);
    k_smax<<<cH * cB * Sq, 256, 0, stream>>>(sc, Sq, Sk, 1.f / sqrtf((float)Sk),
                                             adj, bemb, causal);
    // o = softmax @ v
    gemm(sc, (long long)Sq * Sk, Sk, vb, (long long)Sk * cD, cD, 0, nullptr, 0,
         ob, (long long)Sq * cD, cD, Sq, cD, Sk, 0, cH * cB);
    int tot = cH * cB * Sq * cD;
    k_repack<<<(tot + 255) / 256, 256, 0, stream>>>(ob, ocb, Sq, tot);
    gemm(ocb, 0, cH * cD, g.Wo, 0, cD, 0, g.bo, 0, outp, 0, cD,
         Mq, cD, cH * cD, 0, 1);
  };

  // ---- compound encoders (graph attention with bond-weighted scores) ----
  k_embed3<<<cB * cSC, 128, 0, stream>>>(RE, RC, RP, react.elem, react.charge,
                                         react.posW, react.posb, feat);
  gemm(feat, 0, 3 * cD, react.ffW, 0, cD, 0, react.ffb, 0, cbuf, 0, cD,
       cB * cSC, cD, 3 * cD, 1, 1);
  mha(react.ga, cbuf, cSC, cbuf, cSC, RADJ, react.bond, 0, rx);

  k_embed3<<<cB * cSC, 128, 0, stream>>>(PE, PC, PP, prod.elem, prod.charge,
                                         prod.posW, prod.posb, feat);
  gemm(feat, 0, 3 * cD, prod.ffW, 0, cD, 0, prod.ffb, 0, cbuf, 0, cD,
       cB * cSC, cD, 3 * cD, 1, 1);
  mha(prod.ga, cbuf, cSC, cbuf, cSC, PADJ, prod.bond, 0, px);

  { int tot = cB * cSC * 2 * cD;
    k_cat<<<(tot + 255) / 256, 256, 0, stream>>>(rx, px, cat, tot); }
  gemm(cat, 0, 2 * cD, cW, 0, cD, 0, cb, 0, xb, 0, cD, cB * cSC, cD, 2 * cD, 1, 1);

  // ---- theozyme (causal graph attention) ----
  k_embed2<<<cB * cST, 128, 0, stream>>>(TE, TP, theo.elem, theo.posW, theo.posb, feat);
  gemm(feat, 0, 2 * cD, theo.ffW, 0, cD, 0, theo.ffb, 0, cbuf, 0, cD,
       cB * cST, cD, 2 * cD, 1, 1);
  mha(theo.ga, cbuf, cST, cbuf, cST, TADJ, theo.bond, 1, txb);

  // ---- encoder stack ----
  for (int i = 0; i < cNB; ++i) {
    mha(enc[i].mha, xb, cSC, xb, cSC, nullptr, nullptr, 0, abuf);
    k_addln<<<cB * cSC, 128, 0, stream>>>(abuf, enc[i].g, enc[i].be, xb);
    gemm(xb, 0, cD, enc[i].W1, 0, cDFF, 0, enc[i].b1, 0, ffh, 0, cDFF,
         cB * cSC, cDFF, cD, 1, 1);
    gemm(ffh, 0, cDFF, enc[i].W2, 0, cD, 0, enc[i].b2, 0, abuf, 0, cD,
         cB * cSC, cD, cDFF, 0, 1);
    k_addln<<<cB * cSC, 128, 0, stream>>>(abuf, enc[i].g, enc[i].be, xb);
  }

  // ---- decoder stack ----
  for (int i = 0; i < cNB; ++i) {
    mha(dec[i].smha, txb, cST, txb, cST, nullptr, nullptr, 1, abuf);
    k_addln<<<cB * cST, 128, 0, stream>>>(abuf, dec[i].g, dec[i].be, txb);
    mha(dec[i].cmha, txb, cST, xb, cSC, nullptr, nullptr, 0, abuf);
    k_addln<<<cB * cST, 128, 0, stream>>>(abuf, dec[i].g, dec[i].be, txb);
    gemm(txb, 0, cD, dec[i].W1, 0, cDFF, 0, dec[i].b1, 0, ffh, 0, cDFF,
         cB * cST, cDFF, cD, 1, 1);
    gemm(ffh, 0, cDFF, dec[i].W2, 0, cD, 0, dec[i].b2, 0, abuf, 0, cD,
         cB * cST, cD, cDFF, 0, 1);
    k_addln<<<cB * cST, 128, 0, stream>>>(abuf, dec[i].g, dec[i].be, txb);
  }

  // ---- elem head (softmax over batch axis) ----
  gemm(txb, 0, cD, oo.eW1, 0, cD, 0, oo.eb1, 0, t1, 0, cD, cB * cST, cD, cD, 1, 1);
  gemm(t1, 0, cD, oo.eW2, 0, cELEM, 0, oo.eb2, 0, heb, 0, cELEM,
       cB * cST, cELEM, cD, 0, 1);
  k_elem_softmax<<<(cST * cELEM + 255) / 256, 256, 0, stream>>>(heb, out_elem);

  // ---- pos head ----
  gemm(txb, 0, cD, oo.pW1, 0, cD, 0, oo.pb1, 0, t1, 0, cD, cB * cST, cD, cD, 1, 1);
  gemm(t1, 0, cD, oo.pW2, 0, 3, 0, oo.pb2, 0, out_pos, 0, 3, cB * cST, 3, cD, 0, 1);

  // ---- fused bond head + axis-1 softmax (in-place on d_out) ----
  k_bond<<<dim3(cST / 16, cST, cB), 32, 0, stream>>>(
      txb, oo.b1W, oo.b1b, oo.b2W1, oo.b2b1, oo.b2W2, oo.b2b2, out_bond);
  k_bond_softmax<<<cB * cST * cNBOND, 32, 0, stream>>>(out_bond);
}
